// Encoder_1623497638549
// MI455X (gfx1250) — matrix operand order repaired and checked by reading, not compiled, not run on hardware
//
#include <hip/hip_runtime.h>

// ---------------------------------------------------------------------------
// Types for gfx1250 WMMA (wave32): V_WMMA_F32_16X16X32_BF16
// ---------------------------------------------------------------------------
typedef __attribute__((ext_vector_type(16))) __bf16    v16bf;
typedef __attribute__((ext_vector_type(8)))  __bf16    v8bf;
typedef __attribute__((ext_vector_type(8)))  float     v8f;
typedef __attribute__((ext_vector_type(8)))  unsigned  v8u;

__device__ __forceinline__ __bf16 f2bf(float f) {
  union { float f; unsigned u; } c; c.f = f;
  unsigned r = c.u + 0x7FFFu + ((c.u >> 16) & 1u);   // round-to-nearest-even
  unsigned short h = (unsigned short)(r >> 16);
  return __builtin_bit_cast(__bf16, h);
}
__device__ __forceinline__ float bf2f(__bf16 b) {
  unsigned short h = __builtin_bit_cast(unsigned short, b);
  union { unsigned u; float f; } c; c.u = ((unsigned)h) << 16;
  return c.f;
}
__device__ __forceinline__ void atomicMaxF(float* addr, float val) {
  unsigned* ua = (unsigned*)addr;
  unsigned old = *ua;
  while (true) {
    float f = __builtin_bit_cast(float, old);
    if (f >= val) break;
    unsigned assumed = old;
    old = atomicCAS(ua, assumed, __builtin_bit_cast(unsigned, val));
    if (old == assumed) break;
  }
}
__device__ __forceinline__ v16bf zero16bf() {
  v8u z = {0, 0, 0, 0, 0, 0, 0, 0};
  return __builtin_bit_cast(v16bf, z);
}
__device__ __forceinline__ v16bf join16(v8bf lo, v8bf hi) {
  v16bf r;
#pragma unroll
  for (int i = 0; i < 8; ++i) { r[i] = lo[i]; r[i + 8] = hi[i]; }
  return r;
}

#define N_IMG   2048
#define N_EDGE  32768
#define N_ET    (N_EDGE + N_IMG)   // edges + self loops
#define F_EDGE  32
#define HID     256

// ---------------------------------------------------------------------------
// conv1: Cin=1 direct (0.1% of FLOPs, K=9 too small for WMMA). NHWC bf16 out.
// ---------------------------------------------------------------------------
__global__ void __launch_bounds__(256) conv1_direct(
    const float* __restrict__ x,     // [2048,1,32,32]
    const float* __restrict__ W,     // [64,1,3,3]
    const float* __restrict__ b,     // [64]
    __bf16* __restrict__ out)        // NHWC [2048,32,32,64]
{
  int t   = blockIdx.x * 256 + threadIdx.x;  // == NHWC flat index
  int co  = t & 63;
  int pix = (t >> 6) & 1023;
  int img = t >> 16;
  int oh  = pix >> 5, ow = pix & 31;
  const float* xi = x + img * 1024;
  const float* wf = W + co * 9;
  float s = b[co];
#pragma unroll
  for (int kh = 0; kh < 3; ++kh)
#pragma unroll
    for (int kw = 0; kw < 3; ++kw) {
      int ih = oh + kh - 1, iw = ow + kw - 1;
      if ((unsigned)ih < 32u && (unsigned)iw < 32u)
        s += xi[ih * 32 + iw] * wf[kh * 3 + kw];
    }
  out[t] = f2bf(fmaxf(s, 0.0f));
}

// ---------------------------------------------------------------------------
// Weight prep: OIHW f32 -> [Cout][ (kh*3+kw)*Cin + ci ] bf16  (GEMM B^T)
// ---------------------------------------------------------------------------
__global__ void __launch_bounds__(256) prep_conv_w(
    const float* __restrict__ W, __bf16* __restrict__ Bt, int Cin, int Cout)
{
  int K = 9 * Cin;
  int t = blockIdx.x * 256 + threadIdx.x;
  if (t >= Cout * K) return;
  int co = t / K, k = t - co * K;
  int seg = k / Cin, ci = k - seg * Cin;
  int kh = seg / 3, kw = seg - 3 * kh;
  Bt[t] = f2bf(W[((size_t)(co * Cin + ci) * 3 + kh) * 3 + kw]);
}

// dense [K,N] f32 -> B^T [N,K] bf16
__global__ void __launch_bounds__(256) prep_dense_w(
    const float* __restrict__ W, __bf16* __restrict__ Bt, int K, int N)
{
  int t = blockIdx.x * 256 + threadIdx.x;
  if (t >= K * N) return;
  int n = t / K, k = t - n * K;
  Bt[t] = f2bf(W[(size_t)k * N + n]);
}

// ---------------------------------------------------------------------------
// Implicit-GEMM 3x3 SAME conv, NHWC bf16, fused bias+ReLU, bf16 WMMA f32 acc.
// Block = 256 threads = 8 waves as 4(M) x 2(N); each wave computes 16x32 with
// TWO accumulators sharing ONE A fragment -> 2 WMMAs per K-step per wave.
// Block tile = 64(M) x 64(N).  K loop is tap-outer / channel-inner so the
// (kh,kw) decode, halo bounds test and row pointer are hoisted out of the
// inner loop (no divides in the hot path).  Cin % 32 == 0 for all layers, so
// each 32-wide K slice stays inside one tap.  B tile (64N x 32K, 4KB) staged
// through LDS; M, Cout multiples of 64 -> no tail paths.
// ---------------------------------------------------------------------------
__global__ void __launch_bounds__(256) conv_gemm_wmma(
    const __bf16* __restrict__ act,  // NHWC [2048,H,W,Cin]
    const __bf16* __restrict__ wt,   // [Cout, 9*Cin]
    const float*  __restrict__ bias, // [Cout]
    __bf16* __restrict__ out,        // NHWC [2048,H,W,Cout]
    int H, int W, int Cin, int Cout)
{
  const int K    = 9 * Cin;
  const int lane = threadIdx.x & 31;
  const int wave = threadIdx.x >> 5;
  const int wm = wave & 3;                 // 0..3 along M
  const int wn = wave >> 2;                // 0..1 along N
  const int rowBase = blockIdx.x * 64 + wm * 16;
  const int nBase   = blockIdx.y * 64;

  // per-lane A row -> pixel coordinates (one-time divides)
  const int m   = rowBase + (lane & 15);
  const int HW  = H * W;
  const int img = m / HW;
  const int pix = m - img * HW;
  const int oh  = pix / W;
  const int ow  = pix - oh * W;
  const int kbA = (lane & 16) ? 16 : 0;    // A halves: K 0..15 / 16..31
  const int kbB = (lane & 16) ? 16 : 0;    // B halves: K 0..15 / 16..31

  // B staging: thread t moves 8 bf16 (16B): n = t>>2, kk = (t&3)*8
  const int sN = threadIdx.x >> 2;
  const int sK = (threadIdx.x & 3) << 3;
  const __bf16* wrow = wt + (size_t)(nBase + sN) * K + sK;

  __shared__ __bf16 ldsB[64 * 32];
  __bf16* ldsW = &ldsB[sN * 32 + sK];
  const __bf16* bp0 = &ldsB[((wn << 5) + (lane & 15)) * 32 + kbB];
  const __bf16* bp1 = bp0 + 16 * 32;

  v8f acc0 = {}, acc1 = {};

#pragma unroll
  for (int seg = 0; seg < 9; ++seg) {
    const int kh = seg / 3, kw = seg - 3 * (seg / 3);  // constants after unroll
    const int ih = oh + kh - 1, iw = ow + kw - 1;
    const bool ok = (unsigned)ih < (unsigned)H && (unsigned)iw < (unsigned)W;
    const __bf16* arow = act + ((size_t)(img * H + ih) * W + iw) * Cin + kbA;
    const __bf16* wseg = wrow + (size_t)seg * Cin;

    for (int ci0 = 0; ci0 < Cin; ci0 += 32) {
      *(v8bf*)ldsW = *(const v8bf*)(wseg + ci0);       // stage B tile
      __syncthreads();

      v16bf a;
      if (ok) {
        v8bf lo = *(const v8bf*)(arow + ci0);
        v8bf hi = *(const v8bf*)(arow + ci0 + 8);
        a = join16(lo, hi);
      } else {
        a = zero16bf();                                 // zero halo padding
      }

      v16bf b0 = join16(*(const v8bf*)bp0, *(const v8bf*)(bp0 + 8));
      v16bf b1 = join16(*(const v8bf*)bp1, *(const v8bf*)(bp1 + 8));

      acc0 = __builtin_amdgcn_wmma_f32_16x16x32_bf16(
          false, a, false, b0, (short)0, acc0, false, false);
      acc1 = __builtin_amdgcn_wmma_f32_16x16x32_bf16(
          false, a, false, b1, (short)0, acc1, false, false);
      __syncthreads();
    }
  }

  const int n0  = nBase + (wn << 5) + (lane & 15);
  const float bv0 = bias[n0], bv1 = bias[n0 + 16];
  const int m0 = rowBase + ((lane & 16) ? 8 : 0);
#pragma unroll
  for (int v = 0; v < 8; ++v) {
    out[(size_t)(m0 + v) * Cout + n0]      = f2bf(fmaxf(acc0[v] + bv0, 0.0f));
    out[(size_t)(m0 + v) * Cout + n0 + 16] = f2bf(fmaxf(acc1[v] + bv1, 0.0f));
  }
}

// ---------------------------------------------------------------------------
// Dense GEMM: C[M,N] = A[M,K] @ B[K,N] (+bias, opt relu). A bf16 row-major,
// Bt = B^T bf16 [N,K]. 64x64 block tile, 16x32 per wave, 2 WMMAs / K-step.
// Optional f32 and bf16 outputs. M%64==0, N%64==0, K%32==0 for all uses.
// ---------------------------------------------------------------------------
__global__ void __launch_bounds__(256) gemm_wmma(
    const __bf16* __restrict__ A, const __bf16* __restrict__ Bt,
    const float* __restrict__ bias, __bf16* __restrict__ outBf,
    float* __restrict__ outF, int K, int N, int relu)
{
  const int lane = threadIdx.x & 31;
  const int wave = threadIdx.x >> 5;
  const int wm = wave & 3, wn = wave >> 2;
  const int rowBase = blockIdx.x * 64 + wm * 16;
  const int nBase   = blockIdx.y * 64;
  const int m   = rowBase + (lane & 15);
  const int kbA = (lane & 16) ? 16 : 0;
  const int kbB = (lane & 16) ? 16 : 0;

  const int sN = threadIdx.x >> 2;
  const int sK = (threadIdx.x & 3) << 3;
  const __bf16* wrow = Bt + (size_t)(nBase + sN) * K + sK;

  __shared__ __bf16 ldsB[64 * 32];
  __bf16* ldsW = &ldsB[sN * 32 + sK];
  const __bf16* bp0 = &ldsB[((wn << 5) + (lane & 15)) * 32 + kbB];
  const __bf16* bp1 = bp0 + 16 * 32;

  const __bf16* arow = A + (size_t)m * K + kbA;

  v8f acc0 = {}, acc1 = {};
  for (int k0 = 0; k0 < K; k0 += 32) {
    *(v8bf*)ldsW = *(const v8bf*)(wrow + k0);
    __syncthreads();

    v16bf a  = join16(*(const v8bf*)(arow + k0), *(const v8bf*)(arow + k0 + 8));
    v16bf b0 = join16(*(const v8bf*)bp0, *(const v8bf*)(bp0 + 8));
    v16bf b1 = join16(*(const v8bf*)bp1, *(const v8bf*)(bp1 + 8));

    acc0 = __builtin_amdgcn_wmma_f32_16x16x32_bf16(
        false, a, false, b0, (short)0, acc0, false, false);
    acc1 = __builtin_amdgcn_wmma_f32_16x16x32_bf16(
        false, a, false, b1, (short)0, acc1, false, false);
    __syncthreads();
  }

  const int n0 = nBase + (wn << 5) + (lane & 15);
  const float bv0 = bias ? bias[n0] : 0.0f;
  const float bv1 = bias ? bias[n0 + 16] : 0.0f;
  const int m0 = rowBase + ((lane & 16) ? 8 : 0);
#pragma unroll
  for (int v = 0; v < 8; ++v) {
    float r0 = acc0[v] + bv0;
    float r1 = acc1[v] + bv1;
    if (relu) { r0 = fmaxf(r0, 0.0f); r1 = fmaxf(r1, 0.0f); }
    size_t i0 = (size_t)(m0 + v) * N + n0;
    if (outF)  { outF[i0] = r0;  outF[i0 + 16] = r1; }
    if (outBf) { outBf[i0] = f2bf(r0); outBf[i0 + 16] = f2bf(r1); }
  }
}

// ---------------------------------------------------------------------------
// 2x2 maxpool, NHWC bf16
// ---------------------------------------------------------------------------
__global__ void __launch_bounds__(256) maxpool2(
    const __bf16* __restrict__ in, __bf16* __restrict__ out, int Ho, int C, int total)
{
  int t = blockIdx.x * 256 + threadIdx.x;
  if (t >= total) return;
  int c = t % C;
  int r = t / C;
  int ow = r % Ho; r /= Ho;
  int oh = r % Ho;
  int img = r / Ho;
  int Wi = Ho * 2;
  size_t base = ((size_t)(img * Wi + 2 * oh) * Wi + 2 * ow) * C + c;
  float a0 = bf2f(in[base]);
  float a1 = bf2f(in[base + C]);
  float a2 = bf2f(in[base + (size_t)Wi * C]);
  float a3 = bf2f(in[base + (size_t)Wi * C + C]);
  out[t] = f2bf(fmaxf(fmaxf(a0, a1), fmaxf(a2, a3)));
}

// ---------------------------------------------------------------------------
// GAT helper kernels
// ---------------------------------------------------------------------------
__global__ void __launch_bounds__(256) fill_f32(float* p, int n, float v) {
  int t = blockIdx.x * 256 + threadIdx.x;
  if (t < n) p[t] = v;
}

// column means of edge_attr [32768, 32]
__global__ void __launch_bounds__(256) edge_mean_kernel(
    const float* __restrict__ ea, float* __restrict__ mean)
{
  __shared__ float red[256];
  int col = blockIdx.x;  // 0..31
  float s = 0.f;
  for (int r = threadIdx.x; r < N_EDGE; r += 256) s += ea[(size_t)r * F_EDGE + col];
  red[threadIdx.x] = s; __syncthreads();
  for (int st = 128; st > 0; st >>= 1) {
    if (threadIdx.x < st) red[threadIdx.x] += red[threadIdx.x + st];
    __syncthreads();
  }
  if (threadIdx.x == 0) mean[col] = red[0] * (1.0f / (float)N_EDGE);
}

// v[j] = sum_c We[j,c]*att_edge[c];  vout[32] = dot(mean, v)  (self-loop term)
// (uses (ea @ We) . att_edge == ea . (We @ att_edge))
__global__ void edge_proj_vec(const float* __restrict__ We,
                              const float* __restrict__ att_edge,
                              const float* __restrict__ mean,
                              float* __restrict__ vout)
{
  int j = threadIdx.x;  // blockDim = 32 (one wave)
  float s = 0.f;
  for (int c = 0; c < HID; ++c) s += We[(size_t)j * HID + c] * att_edge[c];
  vout[j] = s;
  __syncthreads();
  if (j == 0) {
    float t = 0.f;
    for (int q = 0; q < F_EDGE; ++q) t += mean[q] * vout[q];
    vout[32] = t;
  }
}

__global__ void __launch_bounds__(256) node_dots(
    const float* __restrict__ hp, const float* __restrict__ as,
    const float* __restrict__ ad, float* __restrict__ a_src,
    float* __restrict__ a_dst)
{
  int i = blockIdx.x * 256 + threadIdx.x;
  if (i >= N_IMG) return;
  const float* h = hp + (size_t)i * HID;
  float s = 0.f, d = 0.f;
  for (int c = 0; c < HID; ++c) { s += h[c] * as[c]; d += h[c] * ad[c]; }
  a_src[i] = s; a_dst[i] = d;
}

__global__ void __launch_bounds__(256) edge_dot_v(
    const float* __restrict__ ea, const float* __restrict__ vbuf,
    float* __restrict__ a_edge)
{
  int e = blockIdx.x * 256 + threadIdx.x;
  if (e >= N_EDGE) return;
  const float* row = ea + (size_t)e * F_EDGE;
  float s = 0.f;
  for (int j = 0; j < F_EDGE; ++j) s += row[j] * vbuf[j];
  a_edge[e] = s;
}

// a = leaky_relu(a_src[src] + a_dst[dst] + a_edge, 0.2); segment-max to amax
__global__ void __launch_bounds__(256) edge_att(
    const int* __restrict__ ei, const float* __restrict__ a_edge,
    const float* __restrict__ vbuf, const float* __restrict__ a_src,
    const float* __restrict__ a_dst, float* __restrict__ aArr,
    float* __restrict__ amax)
{
  int e = blockIdx.x * 256 + threadIdx.x;
  if (e >= N_ET) return;
  int s, d; float ae;
  if (e < N_EDGE) { s = ei[e]; d = ei[N_EDGE + e]; ae = a_edge[e]; }
  else            { s = d = e - N_EDGE;            ae = vbuf[32];  }
  float a = a_src[s] + a_dst[d] + ae;
  a = (a > 0.f) ? a : 0.2f * a;
  aArr[e] = a;
  atomicMaxF(&amax[d], a);
}

__global__ void __launch_bounds__(256) edge_exp(
    const int* __restrict__ ei, const float* __restrict__ aArr,
    const float* __restrict__ amax, float* __restrict__ eArr,
    float* __restrict__ denom)
{
  int e = blockIdx.x * 256 + threadIdx.x;
  if (e >= N_ET) return;
  int d = (e < N_EDGE) ? ei[N_EDGE + e] : (e - N_EDGE);
  float v = __expf(aArr[e] - amax[d]);
  eArr[e] = v;
  atomicAdd(&denom[d], v);
}

// out[dst] += alpha * hp[src]; one thread handles 4 channels of one edge
__global__ void __launch_bounds__(256) edge_agg(
    const int* __restrict__ ei, const float* __restrict__ eArr,
    const float* __restrict__ denom, const float* __restrict__ hp,
    float* __restrict__ agg)
{
  int t = blockIdx.x * 256 + threadIdx.x;
  if (t >= N_ET * 64) return;
  int e  = t >> 6;
  int c0 = (t & 63) << 2;
  int s, d;
  if (e < N_EDGE) { s = ei[e]; d = ei[N_EDGE + e]; }
  else            { s = d = e - N_EDGE; }
  float alpha = eArr[e] / (denom[d] + 1e-16f);
  const float* hs = hp + (size_t)s * HID + c0;
  float* o = agg + (size_t)d * HID + c0;
  atomicAdd(o + 0, alpha * hs[0]);
  atomicAdd(o + 1, alpha * hs[1]);
  atomicAdd(o + 2, alpha * hs[2]);
  atomicAdd(o + 3, alpha * hs[3]);
}

__global__ void __launch_bounds__(256) gat_finalize(
    const float* __restrict__ agg, const float* __restrict__ b,
    float* __restrict__ hout, __bf16* __restrict__ hbf)
{
  int t = blockIdx.x * 256 + threadIdx.x;
  if (t >= N_IMG * HID) return;
  float r = agg[t] + b[t & (HID - 1)];
  hout[t] = r;
  if (hbf) hbf[t] = f2bf(r);
}

// ---------------------------------------------------------------------------
// Host orchestration
// ---------------------------------------------------------------------------
static const int  CIN[13]  = {1, 64, 64, 128, 128, 256, 256, 256, 512, 512, 512, 512, 512};
static const int  COUT[13] = {64, 64, 128, 128, 256, 256, 256, 512, 512, 512, 512, 512, 512};
static const int  HWS[13]  = {32, 32, 16, 16, 8, 8, 8, 4, 4, 4, 2, 2, 2};
static const bool POOLA[13]= {false, true, false, true, false, false, true,
                              false, false, true, false, false, true};

extern "C" void kernel_launch(void* const* d_in, const int* in_sizes, int n_in,
                              void* d_out, int out_size, void* d_ws, size_t ws_size,
                              hipStream_t stream) {
  (void)in_sizes; (void)n_in; (void)out_size; (void)ws_size;

  const float* x          = (const float*)d_in[0];
  const int*   edge_index = (const int*)d_in[1];
  const float* edge_attr  = (const float*)d_in[2];
  const float* convW[13]; const float* convB[13];
  for (int i = 0; i < 13; ++i) {
    convW[i] = (const float*)d_in[3 + 2 * i];
    convB[i] = (const float*)d_in[4 + 2 * i];
  }
  const float* fcW = (const float*)d_in[29];
  const float* fcB = (const float*)d_in[30];
  struct GP { const float *W, *We, *as, *ad, *ae, *b; } gp[3];
  for (int g = 0; g < 3; ++g) {
    int base = 31 + 6 * g;
    gp[g].W  = (const float*)d_in[base + 0];
    gp[g].We = (const float*)d_in[base + 1];
    gp[g].as = (const float*)d_in[base + 2];
    gp[g].ad = (const float*)d_in[base + 3];
    gp[g].ae = (const float*)d_in[base + 4];
    gp[g].b  = (const float*)d_in[base + 5];
  }

  // ---- workspace bump allocator --------------------------------------------
  size_t off = 0;
  auto alloc = [&](size_t bytes) -> char* {
    off = (off + 255) & ~(size_t)255;
    char* p = (char*)d_ws + off;
    off += bytes;
    return p;
  };
  const size_t ACT_ELEMS = (size_t)N_IMG * 32 * 32 * 64;   // largest activation
  __bf16* actA = (__bf16*)alloc(ACT_ELEMS * 2);
  __bf16* actB = (__bf16*)alloc(ACT_ELEMS * 2);
  __bf16* convWt[13];
  for (int i = 1; i < 13; ++i) convWt[i] = (__bf16*)alloc((size_t)COUT[i] * 9 * CIN[i] * 2);
  __bf16* fcWt  = (__bf16*)alloc((size_t)HID * 512 * 2);
  __bf16* gatWt = (__bf16*)alloc((size_t)HID * HID * 2);      // reused per layer
  float*  hbuf  = (float*)alloc((size_t)N_IMG * HID * 4);
  __bf16* hbf   = (__bf16*)alloc((size_t)N_IMG * HID * 2);
  float*  hp    = (float*)alloc((size_t)N_IMG * HID * 4);
  float*  agg   = (float*)alloc((size_t)N_IMG * HID * 4);
  float*  a_srcA= (float*)alloc(N_IMG * 4);
  float*  a_dstA= (float*)alloc(N_IMG * 4);
  float*  amax  = (float*)alloc(N_IMG * 4);
  float*  denom = (float*)alloc(N_IMG * 4);
  float*  a_edgeA = (float*)alloc((size_t)N_ET * 4);
  float*  aArr  = (float*)alloc((size_t)N_ET * 4);
  float*  eArr  = (float*)alloc((size_t)N_ET * 4);
  float*  meanb = (float*)alloc(F_EDGE * 4);
  float*  vbuf  = (float*)alloc(33 * 4);

  // ---- VGG16 feature stack --------------------------------------------------
  __bf16* cur = actA; __bf16* nxt = actB;
  {
    size_t tot = (size_t)N_IMG * 1024 * 64;
    conv1_direct<<<(unsigned)(tot / 256), 256, 0, stream>>>(x, convW[0], convB[0], cur);
  }
  int hw = 32, C = 64;
  for (int i = 1; i < 13; ++i) {
    int Cin = CIN[i], Cout = COUT[i]; hw = HWS[i];
    int Kw = 9 * Cin, totw = Cout * Kw;
    prep_conv_w<<<(totw + 255) / 256, 256, 0, stream>>>(convW[i], convWt[i], Cin, Cout);
    int M = N_IMG * hw * hw;
    dim3 grid(M / 64, Cout / 64);
    conv_gemm_wmma<<<grid, 256, 0, stream>>>(cur, convWt[i], convB[i], nxt, hw, hw, Cin, Cout);
    { __bf16* t = cur; cur = nxt; nxt = t; }
    C = Cout;
    if (POOLA[i]) {
      int ho = hw / 2;
      int tot = N_IMG * ho * ho * C;
      maxpool2<<<(tot + 255) / 256, 256, 0, stream>>>(cur, nxt, ho, C, tot);
      { __bf16* t = cur; cur = nxt; nxt = t; }
      hw = ho;
    }
  }
  // cur now holds [2048, 512] bf16 (1x1 spatial, NHWC == row-major)

  // ---- fc: [2048,512] @ [512,256] + b ---------------------------------------
  prep_dense_w<<<(512 * HID + 255) / 256, 256, 0, stream>>>(fcW, fcWt, 512, HID);
  {
    dim3 grid(N_IMG / 64, HID / 64);
    gemm_wmma<<<grid, 256, 0, stream>>>(cur, fcWt, fcB, hbf, hbuf, 512, HID, 0);
  }

  // ---- shared edge statistics ----------------------------------------------
  edge_mean_kernel<<<F_EDGE, 256, 0, stream>>>(edge_attr, meanb);

  // ---- 3x GATConv -----------------------------------------------------------
  for (int g = 0; g < 3; ++g) {
    prep_dense_w<<<(HID * HID + 255) / 256, 256, 0, stream>>>(gp[g].W, gatWt, HID, HID);
    {
      dim3 grid(N_IMG / 64, HID / 64);
      gemm_wmma<<<grid, 256, 0, stream>>>(hbf, gatWt, nullptr, nullptr, hp, HID, HID, 0);
    }
    edge_proj_vec<<<1, 32, 0, stream>>>(gp[g].We, gp[g].ae, meanb, vbuf);
    node_dots<<<(N_IMG + 255) / 256, 256, 0, stream>>>(hp, gp[g].as, gp[g].ad, a_srcA, a_dstA);
    edge_dot_v<<<(N_EDGE + 255) / 256, 256, 0, stream>>>(edge_attr, vbuf, a_edgeA);

    fill_f32<<<(N_IMG + 255) / 256, 256, 0, stream>>>(amax, N_IMG, -__builtin_inff());
    fill_f32<<<(N_IMG + 255) / 256, 256, 0, stream>>>(denom, N_IMG, 0.0f);
    fill_f32<<<(N_IMG * HID + 255) / 256, 256, 0, stream>>>(agg, N_IMG * HID, 0.0f);

    edge_att<<<(N_ET + 255) / 256, 256, 0, stream>>>(edge_index, a_edgeA, vbuf,
                                                     a_srcA, a_dstA, aArr, amax);
    edge_exp<<<(N_ET + 255) / 256, 256, 0, stream>>>(edge_index, aArr, amax, eArr, denom);
    edge_agg<<<(N_ET * 64 + 255) / 256, 256, 0, stream>>>(edge_index, eArr, denom, hp, agg);

    float* hout = (g == 2) ? (float*)d_out : hbuf;
    gat_finalize<<<(N_IMG * HID + 255) / 256, 256, 0, stream>>>(agg, gp[g].b, hout, hbf);
  }

  // ---- echo edge_index (bit pattern) and edge_attr into d_out ---------------
  float* outF = (float*)d_out;
  hipMemcpyAsync(outF + (size_t)N_IMG * HID, edge_index,
                 (size_t)2 * N_EDGE * sizeof(int), hipMemcpyDeviceToDevice, stream);
  hipMemcpyAsync(outF + (size_t)N_IMG * HID + 2 * N_EDGE, edge_attr,
                 (size_t)N_EDGE * F_EDGE * sizeof(float), hipMemcpyDeviceToDevice, stream);
}
